// EncoderLayer_1494648619697
// MI455X (gfx1250) — matrix operand order repaired and checked by reading, not compiled, not run on hardware
//
#include <hip/hip_runtime.h>
#include <hip/hip_bf16.h>
#include <math.h>

typedef __attribute__((ext_vector_type(16))) __bf16 v16bf;
typedef __attribute__((ext_vector_type(8)))  float  v8f;

#define DMODEL 1024
#define DFF    4096
#define NHEADS 16
#define DK     64
#define SEQL   2048
#define BATCH  2
#define NTOK   (BATCH * SEQL)

// branch-free erf (Abramowitz-Stegun 7.1.26, |err| < 1.5e-7)
__device__ __forceinline__ float erf_poly(float x) {
    float ax = fabsf(x);
    float t  = 1.0f / (1.0f + 0.3275911f * ax);
    float y  = t * (0.254829592f +
               t * (-0.284496736f +
               t * (1.421413741f +
               t * (-1.453152027f +
               t * 1.061405429f))));
    float e = 1.0f - y * __expf(-ax * ax);
    return copysignf(e, x);
}
__device__ __forceinline__ float gelu_exact(float x) {
    return 0.5f * x * (1.0f + erf_poly(x * 0.70710678118654752f));
}

// ---------------- fp32 -> bf16 streaming convert ----------------
__global__ void cvt_f32_bf16(const float* __restrict__ s, __bf16* __restrict__ d, size_t n) {
    size_t i = (size_t)blockIdx.x * blockDim.x + threadIdx.x;
    size_t stride = (size_t)gridDim.x * blockDim.x;
    for (; i < n; i += stride) d[i] = (__bf16)s[i];
}

// ---------------- fp32 -> bf16 transposing convert: out[N][K] = in[K][N] ----------------
// grid (N/32, K/32), 256 threads (32x8)
__global__ __launch_bounds__(256) void cvt_t_f32_bf16(
    const float* __restrict__ in, __bf16* __restrict__ out, int K, int N)
{
    __shared__ float tile[32][33];
    const int bn = blockIdx.x * 32;   // N offset
    const int bk = blockIdx.y * 32;   // K offset
    const int tx = threadIdx.x & 31, ty = threadIdx.x >> 5;
#pragma unroll
    for (int i = 0; i < 32; i += 8)
        tile[ty + i][tx] = in[(size_t)(bk + ty + i) * N + bn + tx];
    __syncthreads();
#pragma unroll
    for (int i = 0; i < 32; i += 8)
        out[(size_t)(bn + ty + i) * K + bk + tx] = (__bf16)tile[tx][ty + i];
}

// ---------------- bf16 WMMA GEMM: C = A[MxK] * Wt^T + bias (Wt given as [N][K]) ----------------
// Block tile 64x128, K-step 32. 8 waves as 2(M) x 4(N); each wave 32x32 (4 accum tiles).
// Ping-pong LDS + one-tile-ahead register staging: one barrier per k-step.
__global__ __launch_bounds__(256) void gemm_bf16_wmma(
    const __bf16* __restrict__ A, const __bf16* __restrict__ Wt,
    const float* __restrict__ bias, float* __restrict__ Cf, __bf16* __restrict__ Cb,
    int M, int N, int K, int act)
{
    __shared__ __bf16 As[2][64][40];     // [buf][m][k]  row stride 80B (16B multiple)
    __shared__ __bf16 Bs[2][128][40];    // [buf][n][k]

    const int tid  = threadIdx.x;
    const int wave = tid >> 5, lane = tid & 31;
    const int hl = lane >> 4, ln = lane & 15;
    const int wm = wave & 1, wn = wave >> 1;
    const int m0 = blockIdx.x * 64, n0 = blockIdx.y * 128;

    v8f acc[2][2];
#pragma unroll
    for (int i = 0; i < 2; ++i)
#pragma unroll
        for (int j = 0; j < 2; ++j)
#pragma unroll
            for (int r = 0; r < 8; ++r) acc[i][j][r] = 0.f;

    const int arow = tid >> 2, acol = (tid & 3) * 8;   // A tile 64x32, 8 bf16/thread
    const int brow = tid >> 1, bcol = (tid & 1) * 16;  // B tile 128x32, 16 bf16/thread

    const __bf16* agp = A  + (size_t)(m0 + arow) * K + acol;
    const __bf16* bgp = Wt + (size_t)(n0 + brow) * K + bcol;

    // prologue: tile 0 -> LDS buf0; tile 1 -> registers
    uint4 ar, br0, br1;
    ar = *(const uint4*)agp;
    { const uint4* bp = (const uint4*)bgp; br0 = bp[0]; br1 = bp[1]; }
    *(uint4*)&As[0][arow][acol] = ar;
    { uint4* bd = (uint4*)&Bs[0][brow][bcol]; bd[0] = br0; bd[1] = br1; }
    if (K > 32) {
        ar = *(const uint4*)(agp + 32);
        const uint4* bp = (const uint4*)(bgp + 32);
        br0 = bp[0]; br1 = bp[1];
    }
    __syncthreads();

    for (int k0 = 0; k0 < K; k0 += 32) {
        const int cur = (k0 >> 5) & 1;
        // commit next tile to the other buffer; kick off loads for tile k0+64
        if (k0 + 32 < K) {
            *(uint4*)&As[cur ^ 1][arow][acol] = ar;
            uint4* bd = (uint4*)&Bs[cur ^ 1][brow][bcol];
            bd[0] = br0; bd[1] = br1;
            if (k0 + 64 < K) {
                ar = *(const uint4*)(agp + k0 + 64);
                const uint4* bp = (const uint4*)(bgp + k0 + 64);
                br0 = bp[0]; br1 = bp[1];
                if (k0 + 96 < K) {
                    __builtin_prefetch(agp + k0 + 96, 0, 1);
                    __builtin_prefetch(bgp + k0 + 96, 0, 1);
                }
            }
        }

        // A fragment: lane m = ln; elem e -> K = hl*16 + e (contig 16-run)
        v16bf afr[2], bfr[2];
#pragma unroll
        for (int ms = 0; ms < 2; ++ms) {
            const int rowA = wm * 32 + ms * 16 + ln;
#pragma unroll
            for (int e = 0; e < 16; ++e) {
                int kk = hl * 16 + e;
                afr[ms][e] = As[cur][rowA][kk];
            }
        }
        // B fragment: lane n = ln; elem e -> K = hl*16 + e (contig 16-run)
#pragma unroll
        for (int ns = 0; ns < 2; ++ns) {
            const int rowB = wn * 32 + ns * 16 + ln;
#pragma unroll
            for (int e = 0; e < 16; ++e)
                bfr[ns][e] = Bs[cur][rowB][hl * 16 + e];
        }
#pragma unroll
        for (int ms = 0; ms < 2; ++ms)
#pragma unroll
            for (int ns = 0; ns < 2; ++ns)
                acc[ms][ns] = __builtin_amdgcn_wmma_f32_16x16x32_bf16(
                    false, afr[ms], false, bfr[ns], (short)0, acc[ms][ns], false, false);
        __syncthreads();   // single barrier per k-step
    }

    // epilogue: C layout row m = r + 8*hl, col n = ln
#pragma unroll
    for (int ms = 0; ms < 2; ++ms)
#pragma unroll
        for (int ns = 0; ns < 2; ++ns)
#pragma unroll
            for (int r = 0; r < 8; ++r) {
                int gm = m0 + wm * 32 + ms * 16 + hl * 8 + r;
                int gn = n0 + wn * 32 + ns * 16 + ln;
                float v = acc[ms][ns][r] + bias[gn];
                if (act) v = gelu_exact(v);
                if (Cf) Cf[(size_t)gm * N + gn] = v;
                if (Cb) Cb[(size_t)gm * N + gn] = (__bf16)v;
            }
}

// ---------------- flash attention (bf16 WMMA, fp32 accum) ----------------
// grid = (SEQL/128, BATCH*NHEADS); 8 waves/block; each wave owns a 16-row Q tile.
// Q/K/V layout: [B, L, H*DK] bf16. Softmax scale folded into Q (1/sqrt(64)=0.125).
// Ping-pong K/V staging: one barrier per 32-key block.
__global__ __launch_bounds__(256) void attn_flash_wmma(
    const __bf16* __restrict__ Q, const __bf16* __restrict__ Kc,
    const __bf16* __restrict__ V, __bf16* __restrict__ O)
{
    __shared__ __bf16 Ks[2][32][72];   // [buf][key][dim] (row stride 144B)
    __shared__ __bf16 Vt[2][64][40];   // [buf][dim][key] transposed (row stride 80B)
    __shared__ __bf16 Ps[8][16][40];   // per-wave P relayout scratch (wave-private)

    const int tid  = threadIdx.x;
    const int wave = tid >> 5, lane = tid & 31;
    const int hl = lane >> 4, ln = lane & 15;
    const int bh = blockIdx.y;
    const int b = bh >> 4, h = bh & 15;
    const int qrow0 = blockIdx.x * 128 + wave * 16;

    // Q fragments (two 32-wide dim chunks), pre-scaled by 0.125
    const __bf16* Qb = Q + ((size_t)(b * SEQL + qrow0)) * DMODEL + h * DK;
    v16bf aQ[2];
#pragma unroll
    for (int c = 0; c < 2; ++c)
#pragma unroll
        for (int e = 0; e < 16; ++e) {
            int kk = hl * 16 + e;
            float qv = (float)Qb[(size_t)ln * DMODEL + c * 32 + kk] * 0.125f;
            aQ[c][e] = (__bf16)qv;
        }

    v8f acc[4];
#pragma unroll
    for (int t = 0; t < 4; ++t)
#pragma unroll
        for (int r = 0; r < 8; ++r) acc[t][r] = 0.f;
    float mrow[8], lrow[8];
#pragma unroll
    for (int r = 0; r < 8; ++r) { mrow[r] = -1e30f; lrow[r] = 0.f; }

    const int srow = tid >> 3;          // key row 0..31
    const int scol = (tid & 7) * 8;     // dim col 0,8,..,56

    const __bf16* Kgp = Kc + ((size_t)(b * SEQL + srow)) * DMODEL + h * DK + scol;
    const __bf16* Vgp = V  + ((size_t)(b * SEQL + srow)) * DMODEL + h * DK + scol;

    // prologue: tile 0 -> LDS buf0; tile 1 -> registers
    uint4 kreg = *(const uint4*)Kgp;
    uint4 vreg = *(const uint4*)Vgp;
    *(uint4*)&Ks[0][srow][scol] = kreg;
    {
        const __bf16* vp = (const __bf16*)&vreg;
#pragma unroll
        for (int j = 0; j < 8; ++j) Vt[0][scol + j][srow] = vp[j];
    }
    kreg = *(const uint4*)(Kgp + (size_t)32 * DMODEL);
    vreg = *(const uint4*)(Vgp + (size_t)32 * DMODEL);
    __syncthreads();

    for (int kb = 0; kb < SEQL; kb += 32) {
        const int cur = (kb >> 5) & 1;
        // commit next K (row-major) / V (transposed) tile to the other buffer
        if (kb + 32 < SEQL) {
            *(uint4*)&Ks[cur ^ 1][srow][scol] = kreg;
            const __bf16* vp = (const __bf16*)&vreg;
#pragma unroll
            for (int j = 0; j < 8; ++j) Vt[cur ^ 1][scol + j][srow] = vp[j];
            if (kb + 64 < SEQL) {
                kreg = *(const uint4*)(Kgp + (size_t)(kb + 64) * DMODEL);
                vreg = *(const uint4*)(Vgp + (size_t)(kb + 64) * DMODEL);
            }
        }

        // S = Q * K^T for 32 keys -> two 16x16 f32 tiles
        v8f s0, s1;
#pragma unroll
        for (int r = 0; r < 8; ++r) { s0[r] = 0.f; s1[r] = 0.f; }
#pragma unroll
        for (int c = 0; c < 2; ++c) {
            v16bf bk0, bk1;
#pragma unroll
            for (int e = 0; e < 16; ++e) {
                int dim = c * 32 + hl * 16 + e;       // contiguous 16-run
                bk0[e] = Ks[cur][ln][dim];            // keys 0..15
                bk1[e] = Ks[cur][16 + ln][dim];       // keys 16..31
            }
            s0 = __builtin_amdgcn_wmma_f32_16x16x32_bf16(false, aQ[c], false, bk0, (short)0, s0, false, false);
            s1 = __builtin_amdgcn_wmma_f32_16x16x32_bf16(false, aQ[c], false, bk1, (short)0, s1, false, false);
        }

        // online softmax; row m = r + 8*hl lives in 16-lane group [hl*16, hl*16+16)
#pragma unroll
        for (int r = 0; r < 8; ++r) {
            float v0 = s0[r], v1 = s1[r];
            float mx = fmaxf(v0, v1);
#pragma unroll
            for (int off = 8; off > 0; off >>= 1) mx = fmaxf(mx, __shfl_xor(mx, off, 32));
            float mnew  = fmaxf(mrow[r], mx);
            float alpha = __expf(mrow[r] - mnew);
            float p0 = __expf(v0 - mnew);
            float p1 = __expf(v1 - mnew);
            float sum = p0 + p1;
#pragma unroll
            for (int off = 8; off > 0; off >>= 1) sum += __shfl_xor(sum, off, 32);
            lrow[r] = lrow[r] * alpha + sum;
            mrow[r] = mnew;
            Ps[wave][hl * 8 + r][ln]      = (__bf16)p0;
            Ps[wave][hl * 8 + r][16 + ln] = (__bf16)p1;
#pragma unroll
            for (int t = 0; t < 4; ++t) acc[t][r] *= alpha;
        }

        // relayout P (C-layout -> A-layout) through per-wave LDS scratch (contig 16-runs)
        v16bf aP;
#pragma unroll
        for (int e = 0; e < 16; ++e) {
            int kk = hl * 16 + e;
            aP[e] = Ps[wave][ln][kk];
        }
        // ctx += P(16x32) * V(32x64): 4 N-tiles; Vt rows give contiguous 16-runs
#pragma unroll
        for (int t = 0; t < 4; ++t) {
            v16bf bv_;
#pragma unroll
            for (int e = 0; e < 16; ++e) bv_[e] = Vt[cur][t * 16 + ln][hl * 16 + e];
            acc[t] = __builtin_amdgcn_wmma_f32_16x16x32_bf16(false, aP, false, bv_, (short)0, acc[t], false, false);
        }
        __syncthreads();   // single barrier per key block
    }

    // finalize: ctx /= l, write bf16 [B, L, H*DK]
    __bf16* Ob = O + ((size_t)(b * SEQL + qrow0)) * DMODEL + h * DK;
#pragma unroll
    for (int r = 0; r < 8; ++r) {
        float inv = 1.0f / lrow[r];
        int m = hl * 8 + r;
#pragma unroll
        for (int t = 0; t < 4; ++t)
            Ob[(size_t)m * DMODEL + t * 16 + ln] = (__bf16)(acc[t][r] * inv);
    }
}

// ---------------- residual add + LayerNorm (one block per token row) ----------------
__global__ __launch_bounds__(256) void add_ln_kernel(
    const float* __restrict__ X, const float* __restrict__ Y,
    const float* __restrict__ g, const float* __restrict__ be,
    float* __restrict__ outf, __bf16* __restrict__ outb)
{
    __shared__ float red[256];
    const int t = threadIdx.x;
    const size_t base = (size_t)blockIdx.x * DMODEL;
    float vals[4];
    float s = 0.f;
#pragma unroll
    for (int i = 0; i < 4; ++i) {
        float v = X[base + t + i * 256] + Y[base + t + i * 256];
        vals[i] = v; s += v;
    }
    red[t] = s; __syncthreads();
    for (int off = 128; off > 0; off >>= 1) { if (t < off) red[t] += red[t + off]; __syncthreads(); }
    const float mu = red[0] * (1.0f / DMODEL);
    __syncthreads();
    float s2 = 0.f;
#pragma unroll
    for (int i = 0; i < 4; ++i) { float d = vals[i] - mu; s2 += d * d; }
    red[t] = s2; __syncthreads();
    for (int off = 128; off > 0; off >>= 1) { if (t < off) red[t] += red[t + off]; __syncthreads(); }
    const float rstd = rsqrtf(red[0] * (1.0f / DMODEL) + 1e-5f);
#pragma unroll
    for (int i = 0; i < 4; ++i) {
        int c = t + i * 256;
        float o = (vals[i] - mu) * rstd * g[c] + be[c];
        if (outf) outf[base + c] = o;
        if (outb) outb[base + c] = (__bf16)o;
    }
}

// ---------------- orchestration ----------------
extern "C" void kernel_launch(void* const* d_in, const int* in_sizes, int n_in,
                              void* d_out, int out_size, void* d_ws, size_t ws_size,
                              hipStream_t stream) {
    (void)in_sizes; (void)n_in; (void)out_size; (void)ws_size;
    const float* x   = (const float*)d_in[0];
    const float* Wq  = (const float*)d_in[1];
    const float* bq  = (const float*)d_in[2];
    const float* Wk  = (const float*)d_in[3];
    const float* bk  = (const float*)d_in[4];
    const float* Wv  = (const float*)d_in[5];
    const float* bv  = (const float*)d_in[6];
    const float* Wo  = (const float*)d_in[7];
    const float* bo  = (const float*)d_in[8];
    const float* W1  = (const float*)d_in[9];
    const float* b1  = (const float*)d_in[10];
    const float* W2  = (const float*)d_in[11];
    const float* b2  = (const float*)d_in[12];
    const float* g1  = (const float*)d_in[13];
    const float* be1 = (const float*)d_in[14];
    const float* g2  = (const float*)d_in[15];
    const float* be2 = (const float*)d_in[16];

    char* p = (char*)d_ws;
    auto take = [&](size_t bytes) -> char* {
        char* r = p; p += (bytes + 255) & ~(size_t)255; return r;
    };
    __bf16* xb   = (__bf16*)take((size_t)NTOK * DMODEL * 2);
    __bf16* wqT  = (__bf16*)take((size_t)DMODEL * DMODEL * 2);  // [N][K]
    __bf16* wkT  = (__bf16*)take((size_t)DMODEL * DMODEL * 2);
    __bf16* wvT  = (__bf16*)take((size_t)DMODEL * DMODEL * 2);
    __bf16* woT  = (__bf16*)take((size_t)DMODEL * DMODEL * 2);
    __bf16* w1T  = (__bf16*)take((size_t)DMODEL * DFF * 2);     // [DFF][DMODEL]
    __bf16* w2T  = (__bf16*)take((size_t)DFF * DMODEL * 2);     // [DMODEL][DFF]
    __bf16* qb_  = (__bf16*)take((size_t)NTOK * DMODEL * 2);
    __bf16* kb_  = (__bf16*)take((size_t)NTOK * DMODEL * 2);
    __bf16* vb_  = (__bf16*)take((size_t)NTOK * DMODEL * 2);
    __bf16* ctxb = (__bf16*)take((size_t)NTOK * DMODEL * 2);
    float*  attnf= (float*) take((size_t)NTOK * DMODEL * 4);
    float*  hf   = (float*) take((size_t)NTOK * DMODEL * 4);
    __bf16* hb   = (__bf16*)take((size_t)NTOK * DMODEL * 2);
    __bf16* ff1b = (__bf16*)take((size_t)NTOK * DFF * 2);
    float*  ff2f = (float*) take((size_t)NTOK * DMODEL * 4);

    // 1) bf16 conversion; weights transposed so GEMM B-operands are [N][K]
    cvt_f32_bf16<<<1024, 256, 0, stream>>>(x, xb, (size_t)NTOK * DMODEL);
    {
        dim3 gdd(DMODEL / 32, DMODEL / 32);
        cvt_t_f32_bf16<<<gdd, 256, 0, stream>>>(Wq, wqT, DMODEL, DMODEL);
        cvt_t_f32_bf16<<<gdd, 256, 0, stream>>>(Wk, wkT, DMODEL, DMODEL);
        cvt_t_f32_bf16<<<gdd, 256, 0, stream>>>(Wv, wvT, DMODEL, DMODEL);
        cvt_t_f32_bf16<<<gdd, 256, 0, stream>>>(Wo, woT, DMODEL, DMODEL);
        dim3 g1d(DFF / 32, DMODEL / 32);
        cvt_t_f32_bf16<<<g1d, 256, 0, stream>>>(W1, w1T, DMODEL, DFF);
        dim3 g2d(DMODEL / 32, DFF / 32);
        cvt_t_f32_bf16<<<g2d, 256, 0, stream>>>(W2, w2T, DFF, DMODEL);
    }

    // 2) QKV projections (bf16 out for attention)
    dim3 gdd(NTOK / 64, DMODEL / 128);
    gemm_bf16_wmma<<<gdd, 256, 0, stream>>>(xb, wqT, bq, nullptr, qb_, NTOK, DMODEL, DMODEL, 0);
    gemm_bf16_wmma<<<gdd, 256, 0, stream>>>(xb, wkT, bk, nullptr, kb_, NTOK, DMODEL, DMODEL, 0);
    gemm_bf16_wmma<<<gdd, 256, 0, stream>>>(xb, wvT, bv, nullptr, vb_, NTOK, DMODEL, DMODEL, 0);

    // 3) flash attention
    dim3 gatt(SEQL / 128, BATCH * NHEADS);
    attn_flash_wmma<<<gatt, 256, 0, stream>>>(qb_, kb_, vb_, ctxb);

    // 4) output projection (fp32 out for residual)
    gemm_bf16_wmma<<<gdd, 256, 0, stream>>>(ctxb, woT, bo, attnf, nullptr, NTOK, DMODEL, DMODEL, 0);

    // 5) h = LN1(x + attn_out) -> fp32 (residual) + bf16 (GEMM input)
    add_ln_kernel<<<NTOK, 256, 0, stream>>>(x, attnf, g1, be1, hf, hb);

    // 6) FFN1 with exact-GELU epilogue (bf16 out)
    dim3 gdf(NTOK / 64, DFF / 128);
    gemm_bf16_wmma<<<gdf, 256, 0, stream>>>(hb, w1T, b1, nullptr, ff1b, NTOK, DFF, DMODEL, 1);

    // 7) FFN2 (fp32 out)
    dim3 gfd(NTOK / 64, DMODEL / 128);
    gemm_bf16_wmma<<<gfd, 256, 0, stream>>>(ff1b, w2T, b2, ff2f, nullptr, NTOK, DMODEL, DFF, 0);

    // 8) out = LN2(h + ff) -> fp32 final output
    add_ln_kernel<<<NTOK, 256, 0, stream>>>(hf, ff2f, g2, be2, (float*)d_out, nullptr);
}
